// MultiHeadAttention_23794118820334
// MI455X (gfx1250) — matrix operand order repaired and checked by reading, not compiled, not run on hardware
//
#include <hip/hip_runtime.h>
#include <hip/hip_bf16.h>

// ---------------------------------------------------------------------------
// MHA forward on gfx1250 (MI455X).
//  - Projections: bf16 WMMA GEMMs (v_wmma_f32_16x16x32_bf16).
//  - Attention: fused logits -> softmax -> P@V kernel; 16 q-rows per block.
//    Q tile staged into LDS via global_load_async_to_lds_b128 (ASYNCcnt),
//    logits in 128KB LDS, bf16 probs in 64KB LDS, attn written to HBM once.
// Shapes fixed by the reference setup: B=2, S=2048, D=1024, H=16, dh=64.
// ---------------------------------------------------------------------------

typedef __bf16 v16bf __attribute__((ext_vector_type(16)));
typedef __bf16 v8bf  __attribute__((ext_vector_type(8)));
typedef float  v8f   __attribute__((ext_vector_type(8)));

static constexpr int  B_  = 2;
static constexpr int  S_  = 2048;
static constexpr int  D_  = 1024;
static constexpr int  H_  = 16;
static constexpr int  DH_ = 64;   // depth per head

// ---- fragment helpers ------------------------------------------------------

__device__ __forceinline__ v8bf cvt_f32x8_to_bf16(const float* p) {
  const float4* q = reinterpret_cast<const float4*>(p);
  float4 a = q[0];
  float4 b = q[1];
  v8bf r;
  r[0] = (__bf16)a.x; r[1] = (__bf16)a.y; r[2] = (__bf16)a.z; r[3] = (__bf16)a.w;
  r[4] = (__bf16)b.x; r[5] = (__bf16)b.y; r[6] = (__bf16)b.z; r[7] = (__bf16)b.w;
  return r;
}

union FragU { v16bf v; struct { v8bf lo; v8bf hi; } s; };

// A-matrix 16x32 bf16 fragment: lane holds row (lane%16); K chunks at
// [hf*8 .. hf*8+7] and [16+hf*8 .. 16+hf*8+7]  (hf = lane/16).
template <typename AT>
__device__ __forceinline__ v16bf load_afrag(const AT* rowp, int kk, int hf) {
  FragU u;
  if constexpr (sizeof(AT) == 2) {
    u.s.lo = *reinterpret_cast<const v8bf*>(rowp + kk + hf * 8);
    u.s.hi = *reinterpret_cast<const v8bf*>(rowp + kk + 16 + hf * 8);
  } else {
    u.s.lo = cvt_f32x8_to_bf16(reinterpret_cast<const float*>(rowp) + kk + hf * 8);
    u.s.hi = cvt_f32x8_to_bf16(reinterpret_cast<const float*>(rowp) + kk + 16 + hf * 8);
  }
  return u.v;
}

// B-matrix 32x16 bf16 fragment from pre-transposed Bt[n,k]: lane holds column
// (lane%16); K chunks at [hf*8 .. hf*8+7] and [16+hf*8 .. 16+hf*8+7].
__device__ __forceinline__ v16bf load_bfrag(const __bf16* colp, int kk, int hf) {
  FragU u;
  u.s.lo = *reinterpret_cast<const v8bf*>(colp + kk + hf * 8);
  u.s.hi = *reinterpret_cast<const v8bf*>(colp + kk + 16 + hf * 8);
  return u.v;
}

__device__ __forceinline__ v8f wmma_bf16(v16bf a, v16bf b, v8f c) {
  return __builtin_amdgcn_wmma_f32_16x16x32_bf16(false, a, false, b, (short)0,
                                                 c, false, false);
}

// ---- unified WMMA GEMM (projections) ---------------------------------------
// C[m,n] = scale * sum_k A[m,k] * Bt[n,k]  (+ bias[n])
// MODE 0: C bf16 row-major   MODE 1: C fp32 row-major
// MODE 3: C bf16 in [B,H,DH,S] layout (V-transposed store)
template <int MODE, typename AT>
__global__ __launch_bounds__(256) void wmma_gemm(
    const AT* __restrict__ A, long lda,
    const __bf16* __restrict__ Bt, long ldb,
    void* __restrict__ Cv, long ldc,
    const float* __restrict__ bias, float scale, int K) {
  const int tid  = threadIdx.x;
  const int wave = tid >> 5;
  const int lane = tid & 31;
  const int lm   = lane & 15;
  const int hf   = lane >> 4;

  const int m0 = blockIdx.y * 128 + wave * 16;
  const int n0 = blockIdx.x * 64;

  v8f acc[4];
#pragma unroll
  for (int t = 0; t < 4; ++t)
#pragma unroll
    for (int j = 0; j < 8; ++j) acc[t][j] = 0.0f;

  const AT* aRow = A + (long)(m0 + lm) * lda;
  const __bf16* bRow0 = Bt + (long)(n0 + lm) * ldb;

  for (int kk = 0; kk < K; kk += 32) {
    // issue all loads for this k-step, then all WMMAs (clause-friendly)
    v16bf af = load_afrag<AT>(aRow, kk, hf);
    v16bf bfr[4];
#pragma unroll
    for (int t = 0; t < 4; ++t)
      bfr[t] = load_bfrag(bRow0 + (long)t * 16 * ldb, kk, hf);
    if (kk + 32 < K) __builtin_prefetch(aRow + kk + 32, 0, 0);
#pragma unroll
    for (int t = 0; t < 4; ++t) acc[t] = wmma_bf16(af, bfr[t], acc[t]);
  }

#pragma unroll
  for (int t = 0; t < 4; ++t) {
#pragma unroll
    for (int j = 0; j < 8; ++j) {
      const int m = m0 + hf * 8 + j;
      const int n = n0 + t * 16 + lm;
      float v = acc[t][j] * scale;
      if (bias != nullptr) v += bias[n];
      if constexpr (MODE == 0) {
        __bf16* C = reinterpret_cast<__bf16*>(Cv);
        C[(long)m * ldc + n] = (__bf16)v;
      } else if constexpr (MODE == 1) {
        float* C = reinterpret_cast<float*>(Cv);
        C[(long)m * ldc + n] = v;
      } else {  // MODE 3: Vt[b, h, dh, s]
        __bf16* C = reinterpret_cast<__bf16*>(Cv);
        const int b = m / S_, s = m % S_;
        const int hh = n / DH_, dh = n % DH_;
        C[(((long)b * H_ + hh) * DH_ + dh) * S_ + s] = (__bf16)v;
      }
    }
  }
}

// ---- fused attention: logits -> softmax -> P@V -----------------------------
// grid = (S/16, B*H), block = 256 (8 waves).
// LDS: Lf = 16x2048 fp32 logits (128KB, later reused for split-K partials),
//      Pb = 16x2048 bf16 probs (64KB), Qt = 16x64 bf16 Q tile (2KB).
__global__ __launch_bounds__(256) void fused_attn(
    const __bf16* __restrict__ Qp,   // [B,S,D] bf16
    const __bf16* __restrict__ Kp,   // [B,S,D] bf16
    const __bf16* __restrict__ Vt,   // [B,H,dh,S] bf16
    float* __restrict__ attn,        // [B,H,S,S] fp32 (output)
    __bf16* __restrict__ Ctx) {      // [B,S,D] bf16
  extern __shared__ char smem[];
  float*  Lf = reinterpret_cast<float*>(smem);                       // 16*S f32
  __bf16* Pb = reinterpret_cast<__bf16*>(smem + 16 * S_ * 4);        // 16*S bf16
  __bf16* Qt = reinterpret_cast<__bf16*>(smem + 16 * S_ * 4 + 16 * S_ * 2);

  const int tid  = threadIdx.x;
  const int wave = tid >> 5;
  const int lane = tid & 31;
  const int lm   = lane & 15;
  const int hf   = lane >> 4;

  const int z = blockIdx.y;        // b*H + h
  const int b = z >> 4;
  const int h = z & 15;
  const int qbase = blockIdx.x * 16;

  // ---- phase 0: async-stage the 16x64 bf16 Q tile into LDS (ASYNCcnt) ----
  if (wave == 0) {
    // low 32 bits of a generic shared pointer == LDS byte offset
    const unsigned qt_lds = (unsigned)(unsigned long long)Qt;
    const int r = lane >> 3;              // 4 rows per issue
    const int cByte = (lane & 7) * 16;    // 8 lanes x 16B cover one 128B row
    const __bf16* qsrc = Qp + ((long)b * S_ + qbase) * D_ + h * DH_;
#pragma unroll
    for (int it = 0; it < 4; ++it) {
      const int row = it * 4 + r;
      unsigned laddr = qt_lds + row * (DH_ * 2) + cByte;
      unsigned long long gaddr =
          (unsigned long long)(const void*)(qsrc + (long)row * D_) + cByte;
      asm volatile("global_load_async_to_lds_b128 %0, %1, off"
                   :: "v"(laddr), "v"(gaddr) : "memory");
    }
    asm volatile("s_wait_asynccnt 0x0" ::: "memory");
  }
  __syncthreads();

  // ---- phase 1: logits tile 16 x S into LDS (scaled) ----
  const __bf16* aRow = Qt + lm * DH_;            // A-fragments from LDS
  const v16bf af0 = load_afrag<__bf16>(aRow, 0, hf);
  const v16bf af1 = load_afrag<__bf16>(aRow, 32, hf);
  const __bf16* Kbase = Kp + (long)b * S_ * D_ + h * DH_;

  for (int t = 0; t < 16; ++t) {
    const int ct = wave + 8 * t;  // column tile 0..127 (key positions ct*16..)
    const __bf16* bRow = Kbase + (long)(ct * 16 + lm) * D_;
    v16bf bf0 = load_bfrag(bRow, 0, hf);
    v16bf bf1 = load_bfrag(bRow, 32, hf);
    v8f acc;
#pragma unroll
    for (int j = 0; j < 8; ++j) acc[j] = 0.0f;
    acc = wmma_bf16(af0, bf0, acc);
    acc = wmma_bf16(af1, bf1, acc);
#pragma unroll
    for (int j = 0; j < 8; ++j)
      Lf[(hf * 8 + j) * S_ + ct * 16 + lm] = acc[j] * 0.125f;
  }
  __syncthreads();

  // ---- phase 2: softmax rows (2 rows per wave), write fp32 attn + bf16 Pb --
#pragma unroll
  for (int rr = 0; rr < 2; ++rr) {
    const int r = wave * 2 + rr;
    const float* row = Lf + (long)r * S_;
    float m = -3.402823466e38f;
    for (int i = lane; i < S_; i += 32) m = fmaxf(m, row[i]);
#pragma unroll
    for (int o = 16; o > 0; o >>= 1) m = fmaxf(m, __shfl_xor(m, o, 32));
    float sum = 0.0f;
    for (int i = lane; i < S_; i += 32) sum += __expf(row[i] - m);
#pragma unroll
    for (int o = 16; o > 0; o >>= 1) sum += __shfl_xor(sum, o, 32);
    const float inv = 1.0f / sum;
    float* gout = attn + ((long)z * S_ + qbase + r) * S_;
    for (int i = lane; i < S_; i += 32) {
      const float p = __expf(row[i] - m) * inv;
      gout[i] = p;                      // coalesced fp32 store (only HBM pass)
      Pb[(long)r * S_ + i] = (__bf16)p; // bf16 copy for the P@V WMMA
    }
  }
  __syncthreads();

  // ---- phase 3: ctx = P @ V_h, split-K across wave pairs ----
  const int t  = wave & 3;   // dh tile: columns t*16 .. t*16+15
  const int kh = wave >> 2;  // K half: [kh*1024, kh*1024+1024)
  const __bf16* vRow = Vt + (long)z * DH_ * S_ + (long)(t * 16 + lm) * S_;
  const __bf16* pRow = Pb + (long)lm * S_;  // A row = lane%16

  v8f acc;
#pragma unroll
  for (int j = 0; j < 8; ++j) acc[j] = 0.0f;
  for (int kk = kh * 1024; kk < kh * 1024 + 1024; kk += 32) {
    v16bf af = load_afrag<__bf16>(pRow, kk, hf);   // ds_load from LDS
    v16bf bf = load_bfrag(vRow, kk, hf);
    __builtin_prefetch(vRow + kk + 32, 0, 0);
    acc = wmma_bf16(af, bf, acc);
  }

  // reduce the two K-halves via LDS (reuse Lf region)
  float* part = Lf;
#pragma unroll
  for (int j = 0; j < 8; ++j)
    part[((kh * 4 + t) * 32 + lane) * 8 + j] = acc[j];
  __syncthreads();

  if (wave < 4) {
    const int tt = wave;
#pragma unroll
    for (int j = 0; j < 8; ++j) {
      const float s = part[((0 * 4 + tt) * 32 + lane) * 8 + j] +
                      part[((1 * 4 + tt) * 32 + lane) * 8 + j];
      Ctx[((long)b * S_ + qbase + hf * 8 + j) * D_ + h * DH_ + tt * 16 + lm] =
          (__bf16)s;
    }
  }
}

// ---- weight transpose + fp32 -> bf16 --------------------------------------
__global__ __launch_bounds__(256) void transpose_cvt_bf16(
    const float* __restrict__ W, __bf16* __restrict__ Wt, int Kdim, int Ndim) {
  const long i = (long)blockIdx.x * blockDim.x + threadIdx.x;
  if (i >= (long)Kdim * Ndim) return;
  const int k = (int)(i / Ndim);
  const int n = (int)(i % Ndim);
  Wt[(long)n * Kdim + k] = (__bf16)W[i];
}

// ---------------------------------------------------------------------------
extern "C" void kernel_launch(void* const* d_in, const int* in_sizes, int n_in,
                              void* d_out, int out_size, void* d_ws, size_t ws_size,
                              hipStream_t stream) {
  (void)in_sizes; (void)n_in; (void)out_size; (void)ws_size;

  const float* q    = (const float*)d_in[0];
  const float* k    = (const float*)d_in[1];
  const float* v    = (const float*)d_in[2];
  const float* wq_w = (const float*)d_in[3];
  const float* wq_b = (const float*)d_in[4];
  const float* wk_w = (const float*)d_in[5];
  const float* wk_b = (const float*)d_in[6];
  const float* wv_w = (const float*)d_in[7];
  const float* wv_b = (const float*)d_in[8];
  const float* wo_w = (const float*)d_in[9];
  const float* wo_b = (const float*)d_in[10];

  float* out  = (float*)d_out;                 // [B,S,D] fp32
  float* attn = out + (long)B_ * S_ * D_;      // [B,H,S,S] fp32

  char* ws = (char*)d_ws;
  size_t off = 0;
  auto take = [&](size_t bytes) {
    void* p = ws + off;
    off += (bytes + 255) & ~(size_t)255;
    return p;
  };
  __bf16* WqT = (__bf16*)take((size_t)D_ * D_ * 2);
  __bf16* WkT = (__bf16*)take((size_t)D_ * D_ * 2);
  __bf16* WvT = (__bf16*)take((size_t)D_ * D_ * 2);
  __bf16* WoT = (__bf16*)take((size_t)D_ * D_ * 2);
  __bf16* Qp  = (__bf16*)take((size_t)B_ * S_ * D_ * 2);       // [B,S,D]
  __bf16* Kp  = (__bf16*)take((size_t)B_ * S_ * D_ * 2);       // [B,S,D]
  __bf16* Vt  = (__bf16*)take((size_t)B_ * H_ * DH_ * S_ * 2); // [B,H,dh,S]
  __bf16* Ctx = (__bf16*)take((size_t)B_ * S_ * D_ * 2);       // [B,S,D]

  const int M = B_ * S_;  // 4096

  // 1) weight transpose+convert
  {
    dim3 g((D_ * D_ + 255) / 256);
    transpose_cvt_bf16<<<g, 256, 0, stream>>>(wq_w, WqT, D_, D_);
    transpose_cvt_bf16<<<g, 256, 0, stream>>>(wk_w, WkT, D_, D_);
    transpose_cvt_bf16<<<g, 256, 0, stream>>>(wv_w, WvT, D_, D_);
    transpose_cvt_bf16<<<g, 256, 0, stream>>>(wo_w, WoT, D_, D_);
  }

  // 2) projections
  {
    dim3 g(D_ / 64, M / 128);
    wmma_gemm<0, float><<<g, 256, 0, stream>>>(q, D_, WqT, D_, Qp, D_, wq_b, 1.0f, D_);
    wmma_gemm<0, float><<<g, 256, 0, stream>>>(k, D_, WkT, D_, Kp, D_, wk_b, 1.0f, D_);
    wmma_gemm<3, float><<<g, 256, 0, stream>>>(v, D_, WvT, D_, Vt, 0, wv_b, 1.0f, D_);
  }

  // 3) fused attention (logits + softmax + P@V), ~194KB dynamic LDS
  {
    const int shmem = 16 * S_ * 4 + 16 * S_ * 2 + 16 * DH_ * 2;  // 198656 B
    static bool attr_done = false;  // idempotent host-side attribute set
    if (!attr_done) {
      (void)hipFuncSetAttribute((const void*)fused_attn,
                                hipFuncAttributeMaxDynamicSharedMemorySize,
                                shmem);
      attr_done = true;
    }
    fused_attn<<<dim3(S_ / 16, B_ * H_), 256, shmem, stream>>>(Qp, Kp, Vt, attn, Ctx);
  }

  // 4) out = Ctx @ Wo + b -> fp32 d_out
  {
    dim3 g(D_ / 64, M / 128);
    wmma_gemm<1, __bf16><<<g, 256, 0, stream>>>(Ctx, D_, WoT, D_, out, D_, wo_b, 1.0f, D_);
  }
}
